// TheAttention_20169166422734
// MI455X (gfx1250) — matrix operand order repaired and checked by reading, not compiled, not run on hardware
//
#include <hip/hip_runtime.h>

#define NB 16
#define T  1024
#define HD 1024
#define UD 1024
#define WPB 8   // waves per block (256 threads, wave32)

typedef __attribute__((ext_vector_type(16))) __bf16 v16bf;
typedef __attribute__((ext_vector_type(8)))  __bf16 v8bf;
typedef __attribute__((ext_vector_type(8)))  float  v8f;

#define WMMA_BF16(a, b, c) \
  __builtin_amdgcn_wmma_f32_16x16x32_bf16(false, (a), false, (b), (short)0, (c), false, false)

static __device__ __forceinline__ v16bf cat8(v8bf lo, v8bf hi) {
  return __builtin_shufflevector(lo, hi, 0,1,2,3,4,5,6,7,8,9,10,11,12,13,14,15);
}

static __device__ __forceinline__ v16bf f32x16_to_bf16(float4 a0, float4 a1, float4 a2, float4 a3) {
  v16bf a;
  a[0]=(__bf16)a0.x; a[1]=(__bf16)a0.y; a[2]=(__bf16)a0.z; a[3]=(__bf16)a0.w;
  a[4]=(__bf16)a1.x; a[5]=(__bf16)a1.y; a[6]=(__bf16)a1.z; a[7]=(__bf16)a1.w;
  a[8]=(__bf16)a2.x; a[9]=(__bf16)a2.y; a[10]=(__bf16)a2.z; a[11]=(__bf16)a2.w;
  a[12]=(__bf16)a3.x; a[13]=(__bf16)a3.y; a[14]=(__bf16)a3.z; a[15]=(__bf16)a3.w;
  return a;
}

// A-fragment (16x32 bf16) from an f32 row: K = c..c+15, c = k0+16*half
static __device__ __forceinline__ v16bf load_a_f32(const float* row, int k0, int half) {
  const float* ap = row + k0 + half * 16;
  return f32x16_to_bf16(*(const float4*)(ap),      *(const float4*)(ap + 4),
                        *(const float4*)(ap + 8),  *(const float4*)(ap + 12));
}

// A-fragment from a bf16 row
static __device__ __forceinline__ v16bf load_a_bf16(const __bf16* row, int k0, int half) {
  v8bf lo = *(const v8bf*)(row + k0 + half * 16);
  v8bf hi = *(const v8bf*)(row + k0 + half * 16 + 8);
  return cat8(lo, hi);
}

// ---------- f32 [R,C] -> bf16 [C,R] transposed convert (32x32 LDS tiles) ----------
__global__ __launch_bounds__(256)
void transpose_w_kernel(const float* __restrict__ W, __bf16* __restrict__ WT) {
  // W: [HD, UD] f32 row-major -> WT: [UD, HD] bf16 row-major
  __shared__ float tile[32][33];
  int bx = blockIdx.x % (UD / 32);  // u tile
  int by = blockIdx.x / (UD / 32);  // h tile
  int tx = threadIdx.x & 31, ty = threadIdx.x >> 5;  // 32 x 8
  #pragma unroll
  for (int r = 0; r < 32; r += 8)
    tile[ty + r][tx] = W[(size_t)(by * 32 + ty + r) * UD + bx * 32 + tx];
  __syncthreads();
  #pragma unroll
  for (int r = 0; r < 32; r += 8)
    WT[(size_t)(bx * 32 + ty + r) * HD + by * 32 + tx] = (__bf16)tile[tx][ty + r];
}

__global__ __launch_bounds__(256)
void transpose_key_kernel(const float* __restrict__ key, __bf16* __restrict__ keyT) {
  // key: [NB, T, HD] f32 -> keyT: [NB, HD, T] bf16
  __shared__ float tile[32][33];
  const int tilesX = HD / 32;
  const int tilesPerBatch = (T / 32) * (HD / 32);
  int b = blockIdx.x / tilesPerBatch;
  int rem = blockIdx.x - b * tilesPerBatch;
  int bx = rem % tilesX;  // h tile
  int by = rem / tilesX;  // t tile
  int tx = threadIdx.x & 31, ty = threadIdx.x >> 5;
  const float* kb = key + (size_t)b * T * HD;
  __bf16* ktb = keyT + (size_t)b * HD * T;
  #pragma unroll
  for (int r = 0; r < 32; r += 8)
    tile[ty + r][tx] = kb[(size_t)(by * 32 + ty + r) * HD + bx * 32 + tx];
  __syncthreads();
  #pragma unroll
  for (int r = 0; r < 32; r += 8)
    ktb[(size_t)(bx * 32 + ty + r) * T + by * 32 + tx] = (__bf16)tile[tx][ty + r];
}

// ---------- Projection: Out_bf16[R,UD] = X[R,HD] @ W + bias, B from WT bf16 [UD,HD] ----------
__global__ __launch_bounds__(32 * WPB)
void proj_kernel(const float* __restrict__ X, const __bf16* __restrict__ WT,
                 const float* __restrict__ bias, __bf16* __restrict__ Out) {
  const int tilesN = UD / 32;                        // 32 (32-wide col tiles)
  int wave = threadIdx.x >> 5;
  int tile = blockIdx.x * WPB + wave;
  int tm = tile / tilesN, tn = tile % tilesN;
  int lane = threadIdx.x & 31, half = lane >> 4, l15 = lane & 15;
  const float*  ar0 = X  + (size_t)(tm * 32 + l15) * HD;
  const float*  ar1 = ar0 + (size_t)16 * HD;
  const __bf16* br0 = WT + (size_t)(tn * 32 + l15) * HD;  // column n -> WT row n
  const __bf16* br1 = br0 + (size_t)16 * HD;
  v8f acc00 = {}, acc01 = {}, acc10 = {}, acc11 = {};
  for (int k0 = 0; k0 < HD; k0 += 32) {
    v16bf a0 = load_a_f32(ar0, k0, half);
    v16bf a1 = load_a_f32(ar1, k0, half);
    v16bf b0 = *(const v16bf*)(br0 + k0 + half * 16);     // B: lane=col, K=k0+16h+i
    v16bf b1 = *(const v16bf*)(br1 + k0 + half * 16);
    acc00 = WMMA_BF16(a0, b0, acc00);
    acc01 = WMMA_BF16(a0, b1, acc01);
    acc10 = WMMA_BF16(a1, b0, acc10);
    acc11 = WMMA_BF16(a1, b1, acc11);
  }
  int n0 = tn * 32 + l15, n1 = n0 + 16;
  float bv0 = bias[n0], bv1 = bias[n1];
  int m0 = tm * 32 + half * 8, m1 = m0 + 16;              // C/D: VGPR r -> M=r+8h
  #pragma unroll
  for (int r = 0; r < 8; ++r) {
    Out[(size_t)(m0 + r) * UD + n0] = (__bf16)(acc00[r] + bv0);
    Out[(size_t)(m0 + r) * UD + n1] = (__bf16)(acc01[r] + bv1);
    Out[(size_t)(m1 + r) * UD + n0] = (__bf16)(acc10[r] + bv0);
    Out[(size_t)(m1 + r) * UD + n1] = (__bf16)(acc11[r] + bv1);
  }
}

// ---------- QK^T / sqrt(U) - mask*1e9 -> logits (f32) ----------
__global__ __launch_bounds__(32 * WPB)
void qk_kernel(const __bf16* __restrict__ Q, const __bf16* __restrict__ K,
               const float* __restrict__ qm, const float* __restrict__ vm,
               float* __restrict__ logits) {
  const int tilesN = T / 32;                         // 32
  const int tilesPerBatch = (T / 32) * (T / 32);     // 1024
  int wave = threadIdx.x >> 5;
  int tile = blockIdx.x * WPB + wave;
  int b = tile / tilesPerBatch;
  int rem = tile - b * tilesPerBatch;
  int tm = rem / tilesN, tn = rem % tilesN;
  int lane = threadIdx.x & 31, half = lane >> 4, l15 = lane & 15;
  const __bf16* ar0 = Q + ((size_t)b * T + tm * 32 + l15) * UD;
  const __bf16* ar1 = ar0 + (size_t)16 * UD;
  const __bf16* br0 = K + ((size_t)b * T + tn * 32 + l15) * UD;  // B col n == K row s
  const __bf16* br1 = br0 + (size_t)16 * UD;
  v8f acc00 = {}, acc01 = {}, acc10 = {}, acc11 = {};
  for (int k0 = 0; k0 < UD; k0 += 32) {
    v16bf a0 = load_a_bf16(ar0, k0, half);
    v16bf a1 = load_a_bf16(ar1, k0, half);
    v16bf b0 = *(const v16bf*)(br0 + k0 + half * 16);
    v16bf b1 = *(const v16bf*)(br1 + k0 + half * 16);
    acc00 = WMMA_BF16(a0, b0, acc00);
    acc01 = WMMA_BF16(a0, b1, acc01);
    acc10 = WMMA_BF16(a1, b0, acc10);
    acc11 = WMMA_BF16(a1, b1, acc11);
  }
  const float inv = 0.03125f;                        // 1/sqrt(1024)
  int s0 = tn * 32 + l15, s1 = s0 + 16;
  float vm0 = vm[b * T + s0], vm1 = vm[b * T + s1];
  int m0 = tm * 32 + half * 8, m1 = m0 + 16;
  float* lg = logits + (size_t)b * T * T;
  #pragma unroll
  for (int r = 0; r < 8; ++r) {
    float qm0 = qm[b * T + m0 + r], qm1 = qm[b * T + m1 + r];
    lg[(size_t)(m0 + r) * T + s0] = acc00[r] * inv - (1.0f - qm0 * vm0) * 1e9f;
    lg[(size_t)(m0 + r) * T + s1] = acc01[r] * inv - (1.0f - qm0 * vm1) * 1e9f;
    lg[(size_t)(m1 + r) * T + s0] = acc10[r] * inv - (1.0f - qm1 * vm0) * 1e9f;
    lg[(size_t)(m1 + r) * T + s1] = acc11[r] * inv - (1.0f - qm1 * vm1) * 1e9f;
  }
}

// ---------- Row softmax, in place ----------
__global__ __launch_bounds__(256)
void softmax_kernel(float* __restrict__ w) {
  __shared__ float sm[256];
  float* p = w + (size_t)blockIdx.x * T;
  int tid = threadIdx.x;
  float lmax = -3.0e38f;
  for (int i = tid; i < T; i += 256) lmax = fmaxf(lmax, p[i]);
  sm[tid] = lmax; __syncthreads();
  for (int s = 128; s > 0; s >>= 1) { if (tid < s) sm[tid] = fmaxf(sm[tid], sm[tid + s]); __syncthreads(); }
  float mx = sm[0]; __syncthreads();
  float lsum = 0.0f;
  for (int i = tid; i < T; i += 256) { float e = __expf(p[i] - mx); p[i] = e; lsum += e; }
  sm[tid] = lsum; __syncthreads();
  for (int s = 128; s > 0; s >>= 1) { if (tid < s) sm[tid] += sm[tid + s]; __syncthreads(); }
  float inv = 1.0f / sm[0];
  for (int i = tid; i < T; i += 256) p[i] *= inv;
}

// ---------- out = weights @ key (B from keyT bf16 [NB,HD,T]), unnormalized ----------
__global__ __launch_bounds__(32 * WPB)
void av_kernel(const float* __restrict__ Wt, const __bf16* __restrict__ keyT,
               float* __restrict__ Out) {
  const int tilesN = HD / 32;
  const int tilesPerBatch = (T / 32) * (HD / 32);
  int wave = threadIdx.x >> 5;
  int tile = blockIdx.x * WPB + wave;
  int b = tile / tilesPerBatch;
  int rem = tile - b * tilesPerBatch;
  int tm = rem / tilesN, tn = rem % tilesN;
  int lane = threadIdx.x & 31, half = lane >> 4, l15 = lane & 15;
  const float*  ar0 = Wt   + ((size_t)b * T + tm * 32 + l15) * T;
  const float*  ar1 = ar0  + (size_t)16 * T;
  const __bf16* br0 = keyT + ((size_t)b * HD + tn * 32 + l15) * T;  // col h -> keyT row h
  const __bf16* br1 = br0  + (size_t)16 * T;
  v8f acc00 = {}, acc01 = {}, acc10 = {}, acc11 = {};
  for (int k0 = 0; k0 < T; k0 += 32) {
    v16bf a0 = load_a_f32(ar0, k0, half);
    v16bf a1 = load_a_f32(ar1, k0, half);
    v16bf b0 = *(const v16bf*)(br0 + k0 + half * 16);
    v16bf b1 = *(const v16bf*)(br1 + k0 + half * 16);
    acc00 = WMMA_BF16(a0, b0, acc00);
    acc01 = WMMA_BF16(a0, b1, acc01);
    acc10 = WMMA_BF16(a1, b0, acc10);
    acc11 = WMMA_BF16(a1, b1, acc11);
  }
  int n0 = tn * 32 + l15, n1 = n0 + 16;
  int m0 = tm * 32 + half * 8, m1 = m0 + 16;
  float* ob = Out + (size_t)b * T * HD;
  #pragma unroll
  for (int r = 0; r < 8; ++r) {
    ob[(size_t)(m0 + r) * HD + n0] = acc00[r];
    ob[(size_t)(m0 + r) * HD + n1] = acc01[r];
    ob[(size_t)(m1 + r) * HD + n0] = acc10[r];
    ob[(size_t)(m1 + r) * HD + n1] = acc11[r];
  }
}

// ---------- LayerNorm over last axis (eps = 1e-3), in place ----------
__global__ __launch_bounds__(256)
void ln_kernel(float* __restrict__ out, const float* __restrict__ gamma,
               const float* __restrict__ beta) {
  __shared__ float sm[256];
  float* p = out + (size_t)blockIdx.x * HD;
  int tid = threadIdx.x;
  float s1 = 0.0f, s2 = 0.0f;
  for (int i = tid; i < HD; i += 256) { float v = p[i]; s1 += v; s2 += v * v; }
  sm[tid] = s1; __syncthreads();
  for (int s = 128; s > 0; s >>= 1) { if (tid < s) sm[tid] += sm[tid + s]; __syncthreads(); }
  float mean = sm[0] * (1.0f / HD); __syncthreads();
  sm[tid] = s2; __syncthreads();
  for (int s = 128; s > 0; s >>= 1) { if (tid < s) sm[tid] += sm[tid + s]; __syncthreads(); }
  float var = sm[0] * (1.0f / HD) - mean * mean;
  float rs = rsqrtf(var + 1e-3f);
  for (int i = tid; i < HD; i += 256)
    p[i] = (p[i] - mean) * rs * gamma[i] + beta[i];
}

extern "C" void kernel_launch(void* const* d_in, const int* in_sizes, int n_in,
                              void* d_out, int out_size, void* d_ws, size_t ws_size,
                              hipStream_t stream) {
  const float* query = (const float*)d_in[0];
  const float* key   = (const float*)d_in[1];
  const float* qmask = (const float*)d_in[2];
  const float* vmask = (const float*)d_in[3];
  const float* Wq_w  = (const float*)d_in[4];
  const float* Wq_b  = (const float*)d_in[5];
  const float* Wk_w  = (const float*)d_in[6];
  const float* Wk_b  = (const float*)d_in[7];
  const float* gamma = (const float*)d_in[8];
  const float* beta  = (const float*)d_in[9];

  float* out_attn = (float*)d_out;                       // [NB,T,HD]
  float* weights  = out_attn + (size_t)NB * T * HD;      // [NB,T,T]

  char* ws = (char*)d_ws;
  __bf16* Qbf  = (__bf16*)ws;                            // [NB*T, UD]  32 MB
  __bf16* Kbf  = (__bf16*)(ws + ((size_t)32 << 20));     // [NB*T, UD]  32 MB
  __bf16* WqT  = (__bf16*)(ws + ((size_t)64 << 20));     // [UD, HD]     2 MB
  __bf16* WkT  = (__bf16*)(ws + ((size_t)66 << 20));     // [UD, HD]     2 MB
  __bf16* keyT = Qbf;   // [NB, HD, T] bf16 — reuses Qbf after qk_kernel consumed it

  transpose_w_kernel<<<(HD / 32) * (UD / 32), 256, 0, stream>>>(Wq_w, WqT);
  transpose_w_kernel<<<(HD / 32) * (UD / 32), 256, 0, stream>>>(Wk_w, WkT);

  const int projTiles = (NB * T / 32) * (UD / 32);       // 16384 waves
  proj_kernel<<<projTiles / WPB, 32 * WPB, 0, stream>>>(query, WqT, Wq_b, Qbf);
  proj_kernel<<<projTiles / WPB, 32 * WPB, 0, stream>>>(key,   WkT, Wk_b, Kbf);

  const int qkTiles = NB * (T / 32) * (T / 32);          // 16384 waves
  qk_kernel<<<qkTiles / WPB, 32 * WPB, 0, stream>>>(Qbf, Kbf, qmask, vmask, weights);

  // After qk, Qbf is dead: overwrite with transposed bf16 key for the AV GEMM.
  transpose_key_kernel<<<NB * (T / 32) * (HD / 32), 256, 0, stream>>>(key, keyT);

  softmax_kernel<<<NB * T, 256, 0, stream>>>(weights);

  const int avTiles = NB * (T / 32) * (HD / 32);         // 16384 waves
  av_kernel<<<avTiles / WPB, 32 * WPB, 0, stream>>>(weights, keyT, out_attn);

  ln_kernel<<<NB * T, 256, 0, stream>>>(out_attn, gamma, beta);
}
